// Shift_tcn_17635135717784
// MI455X (gfx1250) — matrix operand order repaired and checked by reading, not compiled, not run on hardware
//
#include <hip/hip_runtime.h>
#include <hip/hip_bf16.h>

typedef __attribute__((ext_vector_type(16))) _Float16 v16h;
typedef __attribute__((ext_vector_type(8)))  _Float16 v8h;
typedef __attribute__((ext_vector_type(8)))  float    v8f;
typedef int v4i_gcc __attribute__((vector_size(16)));   // matches builtin param

#define AS1 __attribute__((address_space(1)))
#define AS3 __attribute__((address_space(3)))

#if defined(__has_builtin)
#if __has_builtin(__builtin_amdgcn_global_load_async_to_lds_b128) && \
    __has_builtin(__builtin_amdgcn_s_wait_asynccnt)
#define HAVE_ASYNC_LDS 1
#endif
#endif

#define N_IMG 64
#define CIN   128
#define COUT  256
#define T_IN  128
#define V_IN  25
#define T_OUT 64
#define V_OUT 13
#define K_GEMM 1152            // 9 * CIN   (K order: (kt*3+kv)*128 + ci)
#define P_GEMM 53248           // N_IMG * T_OUT * V_OUT
#define KSTEPS 36              // K_GEMM / 32

// ---------------------------------------------------------------- BN1 stats
__global__ __launch_bounds__(256) void bn1_stats(const float* __restrict__ x,
                                                 float* __restrict__ mean1,
                                                 float* __restrict__ rstd1) {
    __shared__ float s1[256];
    __shared__ float s2[256];
    const int c = blockIdx.x;            // 0..127
    float a = 0.f, q = 0.f;
    for (int i = threadIdx.x; i < N_IMG * T_IN * V_IN; i += 256) {
        int n = i / (T_IN * V_IN);
        int r = i - n * (T_IN * V_IN);
        float v = x[((size_t)n * CIN + c) * (T_IN * V_IN) + r];
        a += v; q += v * v;
    }
    s1[threadIdx.x] = a; s2[threadIdx.x] = q;
    __syncthreads();
    for (int off = 128; off > 0; off >>= 1) {
        if (threadIdx.x < off) {
            s1[threadIdx.x] += s1[threadIdx.x + off];
            s2[threadIdx.x] += s2[threadIdx.x + off];
        }
        __syncthreads();
    }
    if (threadIdx.x == 0) {
        const float inv = 1.f / (float)(N_IMG * T_IN * V_IN);
        float m  = s1[0] * inv;
        float var = s2[0] * inv - m * m;
        mean1[c] = m;
        rstd1[c] = rsqrtf(var + 1e-5f);
    }
}

// ---------------------- BN1 + shift_in -> f16 NHWC tensor Hs[n][t][v][ci]
__global__ __launch_bounds__(256) void prep_hs(const float* __restrict__ x,
                                               const float* __restrict__ g1,
                                               const float* __restrict__ b1,
                                               const float* __restrict__ sIn,
                                               const float* __restrict__ mean1,
                                               const float* __restrict__ rstd1,
                                               _Float16* __restrict__ Hs) {
    const int c   = threadIdx.x & (CIN - 1);
    const int pix = blockIdx.x * 2 + (threadIdx.x >> 7);   // 0..204,799
    const int v   = pix % V_IN;
    const int r   = pix / V_IN;
    const int t   = r & (T_IN - 1);
    const int n   = r >> 7;

    const float sc   = sIn[c];
    const float tsrc = (float)t + sc;
    const float fi0  = floorf(tsrc);
    const int   i0   = (int)fi0;
    const int   i1   = i0 + 1;
    const float frac = tsrc - fi0;
    const float w0 = (1.f - frac) * ((i0 >= 0 && i0 < T_IN) ? 1.f : 0.f);
    const float w1 = frac         * ((i1 >= 0 && i1 < T_IN) ? 1.f : 0.f);
    const int i0c = i0 < 0 ? 0 : (i0 > T_IN - 1 ? T_IN - 1 : i0);
    const int i1c = i1 < 0 ? 0 : (i1 > T_IN - 1 ? T_IN - 1 : i1);

    const float m = mean1[c], rs = rstd1[c], gg = g1[c], bb = b1[c];
    const size_t base = ((size_t)n * CIN + c) * T_IN;
    const float x0 = x[(base + i0c) * V_IN + v];
    const float x1 = x[(base + i1c) * V_IN + v];
    const float h0 = (x0 - m) * rs * gg + bb;
    const float h1 = (x1 - m) * rs * gg + bb;
    Hs[(size_t)pix * CIN + c] = (_Float16)(w0 * h0 + w1 * h1);
}

// -------------------- pack weights into WMMA A-fragment layout (ISA 7.12.2)
// K order: k' = (kt*3+kv)*128 + ci.
__global__ __launch_bounds__(256) void pack_w(const float* __restrict__ w,
                                              _Float16* __restrict__ Wfrag) {
    const int gid = blockIdx.x * 256 + threadIdx.x;  // 294,912 total (exact)
    const int e   = gid & 15;
    const int l   = (gid >> 4) & 31;
    const int blk = gid >> 9;           // 0..575
    const int kb  = blk % KSTEPS;
    const int ct  = blk / KSTEPS;       // 0..15

    const int vv  = e >> 1;
    const int bit = e & 1;
    const int k   = kb * 32 + (vv >> 2) * 16 + (l >> 4) * 8 + (vv & 3) * 2 + bit;
    const int co  = ct * 16 + (l & 15);
    const int ci  = k & (CIN - 1);
    const int g   = k >> 7;             // kt*3+kv, 0..8
    const int kt  = g / 3;
    const int kv  = g - kt * 3;
    Wfrag[gid] = (_Float16)w[(((size_t)co * CIN + ci) * 3 + kt) * 3 + kv];
}

// ------------------------------------- implicit GEMM conv + bias + ReLU
// Workgroup tile: 64 co x 128 pixels; 8 wave32 waves, 4 WMMA tiles each.
// Double-buffered LDS B tiles: async fill of tile i+1 overlaps WMMAs on i.
__global__ __launch_bounds__(256) void gemm_conv(const _Float16* __restrict__ Wfrag,
                                                 const _Float16* __restrict__ Hs,
                                                 const float* __restrict__ bias,
                                                 float* __restrict__ Y) {
    __shared__ _Float16 Blds[2][128 * 40];

    const int lane = threadIdx.x & 31;
    const int wave = threadIdx.x >> 5;      // 8 wave32 waves
    const int mw   = wave & 3;              // co 16-tile within 64
    const int nh   = wave >> 2;             // pixel 64-half within 128
    const int pBase = blockIdx.x * 128;

    // A pointer: contiguous 32 B per lane per K-step
    const _Float16* Aptr =
        Wfrag + ((size_t)(blockIdx.y * 4 + mw) * KSTEPS) * 512 + lane * 16;

    // ---- B-fill coordinates: one 32 B copy (16 consecutive channels) each
    const int pix_l = threadIdx.x >> 1;          // 0..127
    const int k0    = (threadIdx.x & 1) << 4;    // 0 or 16
    const int p_f   = pBase + pix_l;
    const int vo_f  = p_f % V_OUT;
    const int tq    = p_f / V_OUT;
    const int to_f  = tq & (T_OUT - 1);
    const int ni_f  = tq >> 6;
    const int t2b   = 2 * to_f - 1;
    const int v2b   = 2 * vo_f - 1;
    const int ldsOff = pix_l * 40 + k0;

    // fragment-read coords (B lane layout: N = lane%16, K-half = lane/16)
    const int rowBase = (nh << 6) + (lane & 15);
    const int kbase   = (lane >> 4) << 3;

    // stage one 32-k B tile (uniform predicate per pixel per tap)
    auto fill = [&](int kb, int buf) {
        const int g   = kb >> 7;            // kt*3+kv (constant per step)
        const int cb0 = kb & (CIN - 1);
        const int dt  = g / 3;
        const int dv  = g - dt * 3;
        const int t   = t2b + dt;
        const int v   = v2b + dv;
        _Float16* lp = &Blds[buf][ldsOff];
        if ((unsigned)t < (unsigned)T_IN && (unsigned)v < (unsigned)V_IN) {
            const _Float16* gp =
                Hs + (((size_t)ni_f * T_IN + t) * V_IN + v) * CIN + cb0 + k0;
#ifdef HAVE_ASYNC_LDS
            __builtin_amdgcn_global_load_async_to_lds_b128(
                (AS1 v4i_gcc*)gp, (AS3 v4i_gcc*)lp, 0, 0);
            __builtin_amdgcn_global_load_async_to_lds_b128(
                (AS1 v4i_gcc*)(gp + 8), (AS3 v4i_gcc*)(lp + 8), 0, 0);
#else
            v8h d0 = *(const v8h*)gp;
            v8h d1 = *(const v8h*)(gp + 8);
            *(v8h*)lp       = d0;
            *(v8h*)(lp + 8) = d1;
#endif
        } else {
            v8h z = {};
            *(v8h*)lp       = z;
            *(v8h*)(lp + 8) = z;
        }
    };

    v8f acc[4] = {v8f{}, v8f{}, v8f{}, v8f{}};

    fill(0, 0);                                 // prologue: tile 0
    for (int i = 0; i < KSTEPS; ++i) {
        const int cur = i & 1;
#ifdef HAVE_ASYNC_LDS
        __builtin_amdgcn_s_wait_asynccnt(0);    // my fills for tile i done
#endif
        __syncthreads();                        // everyone's fills done / reads done
        if (i + 1 < KSTEPS)
            fill((i + 1) * 32, cur ^ 1);        // overlap next fill with compute

        union { v16h v; v8h h[2]; } a;
        a.v = *(const v16h*)Aptr;
        __builtin_prefetch(Aptr + 512, 0, 1);   // next A block
        Aptr += 512;

        #pragma unroll
        for (int t2 = 0; t2 < 4; ++t2) {
            union { v16h v; v8h h[2]; } b;
            const int row = rowBase + (t2 << 4);
            b.h[0] = *(const v8h*)&Blds[cur][row * 40 + kbase];
            b.h[1] = *(const v8h*)&Blds[cur][row * 40 + kbase + 16];
            acc[t2] = __builtin_amdgcn_wmma_f32_16x16x32_f16(
                false, a.v, false, b.v, (short)0, acc[t2], false, false);
        }
    }

    // ---- epilogue: bias + ReLU, scatter to Y (NCHW f32)
    const int co0 = blockIdx.y * 64 + mw * 16 + ((lane >> 4) << 3);
    #pragma unroll
    for (int t2 = 0; t2 < 4; ++t2) {
        const int p  = pBase + rowBase + (t2 << 4);
        const int vo = p % V_OUT;
        const int tp = p / V_OUT;
        const int to = tp & (T_OUT - 1);
        const int ni = tp >> 6;
        #pragma unroll
        for (int r = 0; r < 8; ++r) {
            const int co = co0 + r;
            float o = acc[t2][r] + bias[co];
            o = fmaxf(o, 0.f);
            Y[(((size_t)ni * COUT + co) * T_OUT + to) * V_OUT + vo] = o;
        }
    }
}

// ----------------------------------------- BN2 stats over shift_out(relu(Y))
__global__ __launch_bounds__(256) void bn2_stats(const float* __restrict__ Y,
                                                 const float* __restrict__ sOut,
                                                 float* __restrict__ mean2,
                                                 float* __restrict__ rstd2) {
    __shared__ float s1[256];
    __shared__ float s2[256];
    const int c = blockIdx.x;             // 0..255
    const float sc = sOut[c];
    float a = 0.f, q = 0.f;
    for (int i = threadIdx.x; i < N_IMG * T_OUT * V_OUT; i += 256) {
        const int vo = i % V_OUT;
        const int r  = i / V_OUT;
        const int to = r & (T_OUT - 1);
        const int n  = r >> 6;
        const float tsrc = (float)to + sc;
        const float fi0  = floorf(tsrc);
        const int   i0   = (int)fi0;
        const int   i1   = i0 + 1;
        const float frac = tsrc - fi0;
        const float w0 = (1.f - frac) * ((i0 >= 0 && i0 < T_OUT) ? 1.f : 0.f);
        const float w1 = frac         * ((i1 >= 0 && i1 < T_OUT) ? 1.f : 0.f);
        const int i0c = i0 < 0 ? 0 : (i0 > T_OUT - 1 ? T_OUT - 1 : i0);
        const int i1c = i1 < 0 ? 0 : (i1 > T_OUT - 1 ? T_OUT - 1 : i1);
        const size_t base = ((size_t)n * COUT + c) * T_OUT;
        const float z = w0 * Y[(base + i0c) * V_OUT + vo] +
                        w1 * Y[(base + i1c) * V_OUT + vo];
        a += z; q += z * z;
    }
    s1[threadIdx.x] = a; s2[threadIdx.x] = q;
    __syncthreads();
    for (int off = 128; off > 0; off >>= 1) {
        if (threadIdx.x < off) {
            s1[threadIdx.x] += s1[threadIdx.x + off];
            s2[threadIdx.x] += s2[threadIdx.x + off];
        }
        __syncthreads();
    }
    if (threadIdx.x == 0) {
        const float inv = 1.f / (float)(N_IMG * T_OUT * V_OUT);
        float m   = s1[0] * inv;
        float var = s2[0] * inv - m * m;
        mean2[c] = m;
        rstd2[c] = rsqrtf(var + 1e-5f);
    }
}

// -------------------------------------------- shift_out + BN2 -> output
__global__ __launch_bounds__(256) void finalize(const float* __restrict__ Y,
                                                const float* __restrict__ sOut,
                                                const float* __restrict__ mean2,
                                                const float* __restrict__ rstd2,
                                                const float* __restrict__ g2,
                                                const float* __restrict__ b2,
                                                float* __restrict__ out) {
    const int idx = blockIdx.x * 256 + threadIdx.x;  // 13,631,488 total (exact)
    const int vo = idx % V_OUT;
    const int r  = idx / V_OUT;
    const int to = r & (T_OUT - 1);
    const int c  = (r >> 6) & (COUT - 1);
    const int n  = r >> 14;

    const float sc   = sOut[c];
    const float tsrc = (float)to + sc;
    const float fi0  = floorf(tsrc);
    const int   i0   = (int)fi0;
    const int   i1   = i0 + 1;
    const float frac = tsrc - fi0;
    const float w0 = (1.f - frac) * ((i0 >= 0 && i0 < T_OUT) ? 1.f : 0.f);
    const float w1 = frac         * ((i1 >= 0 && i1 < T_OUT) ? 1.f : 0.f);
    const int i0c = i0 < 0 ? 0 : (i0 > T_OUT - 1 ? T_OUT - 1 : i0);
    const int i1c = i1 < 0 ? 0 : (i1 > T_OUT - 1 ? T_OUT - 1 : i1);

    const size_t base = ((size_t)n * COUT + c) * T_OUT;
    const float z = w0 * Y[(base + i0c) * V_OUT + vo] +
                    w1 * Y[(base + i1c) * V_OUT + vo];
    out[idx] = (z - mean2[c]) * rstd2[c] * g2[c] + b2[c];
}

// ---------------------------------------------------------------- launcher
extern "C" void kernel_launch(void* const* d_in, const int* in_sizes, int n_in,
                              void* d_out, int out_size, void* d_ws, size_t ws_size,
                              hipStream_t stream) {
    (void)in_sizes; (void)n_in; (void)out_size; (void)ws_size;
    const float* x    = (const float*)d_in[0];
    const float* g1   = (const float*)d_in[1];
    const float* b1   = (const float*)d_in[2];
    const float* sIn  = (const float*)d_in[3];
    const float* cw   = (const float*)d_in[4];
    const float* cb   = (const float*)d_in[5];
    const float* sOut = (const float*)d_in[6];
    const float* g2   = (const float*)d_in[7];
    const float* b2   = (const float*)d_in[8];
    float* out = (float*)d_out;

    char* ws = (char*)d_ws;
    float*     mean1 = (float*)(ws + 0);
    float*     rstd1 = (float*)(ws + 512);
    float*     mean2 = (float*)(ws + 1024);
    float*     rstd2 = (float*)(ws + 2048);
    _Float16*  Wfrag = (_Float16*)(ws + 4096);                 //   589,824 B
    _Float16*  Hs    = (_Float16*)(ws + 593920);               // 52,428,800 B
    float*     Y     = (float*)(ws + 53022720);                // 54,525,952 B

    bn1_stats<<<128, 256, 0, stream>>>(x, mean1, rstd1);
    prep_hs<<<102400, 256, 0, stream>>>(x, g1, b1, sIn, mean1, rstd1, Hs);
    pack_w<<<1152, 256, 0, stream>>>(cw, Wfrag);
    gemm_conv<<<dim3(P_GEMM / 128, COUT / 64), 256, 0, stream>>>(Wfrag, Hs, cb, Y);
    bn2_stats<<<256, 256, 0, stream>>>(Y, sOut, mean2, rstd2);
    finalize<<<53248, 256, 0, stream>>>(Y, sOut, mean2, rstd2, g2, b2, out);
}
